// GPT_SparseTransformerBlock_88201448391068
// MI455X (gfx1250) — matrix operand order repaired and checked by reading, not compiled, not run on hardware
//
#include <hip/hip_runtime.h>
#include <hip/hip_bf16.h>

// ---------------------------------------------------------------------------
// GPT sparse-transformer block for gfx1250 (MI455X), wave32 + WMMA.
// GEMMs: v_wmma_f32_16x16x32_f16 (f16 in, f32 accumulate), double-buffered
// LDS tiles fed by GLOBAL_LOAD_ASYNC_TO_LDS_B128 (ASYNCcnt pipeline) so
// global fetch of tile k+1 overlaps WMMA compute of tile k.
// ---------------------------------------------------------------------------

typedef __attribute__((ext_vector_type(16))) _Float16 v16h;
typedef __attribute__((ext_vector_type(8)))  _Float16 v8h;
typedef __attribute__((ext_vector_type(8)))  float    v8f;

#define D_MODEL 1024
#define SEQ     2048
#define BATCH   2
#define ROWS    (BATCH * SEQ)   // 4096
#define HIDDEN  4096
#define NHEADS  16
#define HDIM    64
#define BLK     16

#define KT   64   // K-tile depth (2 WMMA k-steps per LDS refill)
#define LDT2 72   // LDS row stride in halves: 144B, multiple of 16 -> aligned b128

#if defined(__gfx1250__)
#define USE_ASYNC_LDS 1
#else
#define USE_ASYNC_LDS 0
#endif

// ---------------------------------------------------------------------------
// fp32 [K][N]  ->  f16 [N][K]   (weight convert + transpose, done once/launch)
// ---------------------------------------------------------------------------
__global__ void cvt_transpose_kernel(const float* __restrict__ w,
                                     _Float16* __restrict__ wt,
                                     int K, int N) {
    int idx = blockIdx.x * 256 + threadIdx.x;      // index into output [N][K]
    int total = K * N;
    if (idx < total) {
        int n = idx / K;
        int k = idx - n * K;
        wt[idx] = (_Float16)w[(size_t)k * N + n];
    }
}

// ---------------------------------------------------------------------------
// LayerNorm over last dim (D=1024), fp32 in -> f16 out. One block per row.
// ---------------------------------------------------------------------------
__global__ void ln_kernel(const float* __restrict__ x,
                          const float* __restrict__ g,
                          const float* __restrict__ b,
                          _Float16* __restrict__ out) {
    __shared__ float red[256];
    const int row = blockIdx.x;
    const int tid = threadIdx.x;
    const float* xr = x + (size_t)row * D_MODEL;

    float vals[4];
    float s = 0.f;
#pragma unroll
    for (int i = 0; i < 4; ++i) { vals[i] = xr[tid + i * 256]; s += vals[i]; }

    red[tid] = s; __syncthreads();
    for (int off = 128; off > 0; off >>= 1) {
        if (tid < off) red[tid] += red[tid + off];
        __syncthreads();
    }
    const float mean = red[0] * (1.0f / 1024.0f);
    __syncthreads();

    float s2 = 0.f;
#pragma unroll
    for (int i = 0; i < 4; ++i) { float d = vals[i] - mean; s2 += d * d; }
    red[tid] = s2; __syncthreads();
    for (int off = 128; off > 0; off >>= 1) {
        if (tid < off) red[tid] += red[tid + off];
        __syncthreads();
    }
    const float var = red[0] * (1.0f / 1024.0f);
    const float inv = rsqrtf(var + 1e-5f);

    _Float16* orow = out + (size_t)row * D_MODEL;
#pragma unroll
    for (int i = 0; i < 4; ++i) {
        int c = tid + i * 256;
        orow[c] = (_Float16)((vals[i] - mean) * inv * g[c] + b[c]);
    }
}

// ---------------------------------------------------------------------------
// WMMA GEMM:  C[M][N] = A[M][K](f16,row) @ Bt[N][K](f16,row == B col-major)
//             (+bias) (+GELU) (+fp32 residual) -> fp32 or f16 out.
// 128x128 block tile, K-tile 64, 8 waves (2x4 WMMA tiles / wave),
// 16 WMMAs per barrier pair. Double-buffered LDS; tile k+1 streams in via
// GLOBAL_LOAD_ASYNC_TO_LDS_B128 while tile k is consumed by WMMA.
// ---------------------------------------------------------------------------
template <bool HAS_BIAS, bool DO_GELU, bool HAS_RES, bool OUT_HALF>
__global__ void __launch_bounds__(256)
gemm_wmma_kernel(const _Float16* __restrict__ A,
                 const _Float16* __restrict__ Bt,
                 const float* __restrict__ bias,
                 const float* __restrict__ residual,
                 float* __restrict__ outF,
                 _Float16* __restrict__ outH,
                 int M, int N, int K) {
    __shared__ __align__(16) _Float16 Alds[2][128 * LDT2];
    __shared__ __align__(16) _Float16 Blds[2][128 * LDT2];

    const int tid  = threadIdx.x;
    const int lane = tid & 31;
    const int wave = tid >> 5;
    const int wm   = wave >> 1;          // 0..3 : 32-row slab
    const int wn   = wave & 1;           // 0..1 : 64-col slab
    const int lm   = lane & 15;          // row/col within 16-tile
    const int lh   = lane >> 4;          // half-wave selector
    const int bm   = blockIdx.y;
    const int bn   = blockIdx.x;

    // copy-slot decomposition: 128 rows x 8 x 16B segments, 4 rows/thread
    const int seg   = tid & 7;           // 16B segment within a 64-half row
    const int rbase = tid >> 3;          // rows rbase, rbase+32, +64, +96

    v8f acc[2][4] = {};

    // ---- 2 WMMA k-steps from one LDS tile; fragments per ISA 7.12.2 ----
    auto compute_tile = [&](const _Float16* Ab, const _Float16* Bb) {
#pragma unroll
        for (int kk = 0; kk < KT; kk += 32) {
            v16h af[2], bf[4];
#pragma unroll
            for (int mt = 0; mt < 2; ++mt) {
                const _Float16* ap = Ab + (wm * 32 + mt * 16 + lm) * LDT2 + kk + 8 * lh;
                v8h alo = *(const v8h*)ap;          // K = kk+8*lh .. +7
                v8h ahi = *(const v8h*)(ap + 16);   // K = kk+16+8*lh .. +7
                af[mt] = (v16h)__builtin_shufflevector(alo, ahi,
                            0,1,2,3,4,5,6,7,8,9,10,11,12,13,14,15);
            }
#pragma unroll
            for (int nt = 0; nt < 4; ++nt) {
                const _Float16* bp = Bb + (wn * 64 + nt * 16 + lm) * LDT2 + kk + 8 * lh;
                v8h blo = *(const v8h*)bp;          // K = kk+8*lh .. +7
                v8h bhi = *(const v8h*)(bp + 16);   // K = kk+16+8*lh .. +7
                bf[nt] = (v16h)__builtin_shufflevector(blo, bhi,
                            0,1,2,3,4,5,6,7,8,9,10,11,12,13,14,15);
            }
#pragma unroll
            for (int mt = 0; mt < 2; ++mt)
#pragma unroll
                for (int nt = 0; nt < 4; ++nt)
                    acc[mt][nt] = __builtin_amdgcn_wmma_f32_16x16x32_f16(
                        false, af[mt], false, bf[nt],
                        (short)0, acc[mt][nt], false, false);
        }
    };

#if USE_ASYNC_LDS
    // 8 async b128 ops per thread per tile (4 rows x {A,B}); no staging VGPRs.
    auto issue_tile = [&](int k0, int buf) {
#pragma unroll
        for (int it = 0; it < 4; ++it) {
            const int r = rbase + it * 32;
            const unsigned goffA =
                (unsigned)(((size_t)(bm * 128 + r) * K + k0 + seg * 8) * 2);
            const unsigned goffB =
                (unsigned)(((size_t)(bn * 128 + r) * K + k0 + seg * 8) * 2);
            // generic LDS pointer low 32 bits == LDS byte offset (ISA 10.2)
            const unsigned la = (unsigned)(unsigned long long)(uintptr_t)
                &Alds[buf][r * LDT2 + seg * 8];
            const unsigned lb = (unsigned)(unsigned long long)(uintptr_t)
                &Blds[buf][r * LDT2 + seg * 8];
            asm volatile("global_load_async_to_lds_b128 %0, %1, %2"
                         :: "v"(la), "v"(goffA), "s"(A) : "memory");
            asm volatile("global_load_async_to_lds_b128 %0, %1, %2"
                         :: "v"(lb), "v"(goffB), "s"(Bt) : "memory");
        }
    };

    issue_tile(0, 0);
    int buf = 0;
    for (int k0 = 0; k0 < K; k0 += KT) {
        const bool more = (k0 + KT) < K;          // uniform branch
        if (more) {
            issue_tile(k0 + KT, buf ^ 1);
            // async loads complete in order: cnt<=8 => tile k fully in LDS
            asm volatile("s_wait_asynccnt 0x8" ::: "memory");
        } else {
            asm volatile("s_wait_asynccnt 0x0" ::: "memory");
        }
        __syncthreads();                          // tile k visible to all waves
        compute_tile(&Alds[buf][0], &Blds[buf][0]);
        __syncthreads();                          // done reading before overwrite
        buf ^= 1;
    }
#else
    for (int k0 = 0; k0 < K; k0 += KT) {
#pragma unroll
        for (int it = 0; it < 4; ++it) {
            const int r = rbase + it * 32;
            *(uint4*)&Alds[0][r * LDT2 + seg * 8] =
                *(const uint4*)(A + (size_t)(bm * 128 + r) * K + k0 + seg * 8);
            *(uint4*)&Blds[0][r * LDT2 + seg * 8] =
                *(const uint4*)(Bt + (size_t)(bn * 128 + r) * K + k0 + seg * 8);
        }
        __syncthreads();
        compute_tile(&Alds[0][0], &Blds[0][0]);
        __syncthreads();
    }
#endif

    // ---- epilogue (fully branch-free via template flags) ----
#pragma unroll
    for (int mt = 0; mt < 2; ++mt) {
#pragma unroll
        for (int nt = 0; nt < 4; ++nt) {
            const int col = bn * 128 + wn * 64 + nt * 16 + lm;
            float bv = 0.0f;
            if (HAS_BIAS) bv = bias[col];
#pragma unroll
            for (int r = 0; r < 8; ++r) {
                const int row = bm * 128 + wm * 32 + mt * 16 + r + 8 * lh;
                float val = acc[mt][nt][r] + bv;
                if (DO_GELU)
                    val = 0.5f * val * (1.0f + erff(val * 0.70710678118654752f));
                const size_t idx = (size_t)row * N + col;
                if (HAS_RES) val += residual[idx];
                if (OUT_HALF) outH[idx] = (_Float16)val;
                else          outF[idx] = val;
            }
        }
    }
}

// ---------------------------------------------------------------------------
// Sparse attention (fixed pattern, causal): one wave per (b,h,query).
// Keys = summary positions (j % 16 == 15, j < block start) + local block.
// Online softmax; lane holds head dims {lane, lane+32}.
// ---------------------------------------------------------------------------
__global__ void __launch_bounds__(256)
attn_kernel(const _Float16* __restrict__ q,
            const _Float16* __restrict__ k,
            const _Float16* __restrict__ v,
            _Float16* __restrict__ ctx) {
    const int wid  = blockIdx.x * 8 + (threadIdx.x >> 5);
    const int lane = threadIdx.x & 31;
    const int i = wid & (SEQ - 1);          // query position
    const int h = (wid >> 11) & (NHEADS - 1);
    const int b = wid >> 15;

    const size_t base = ((size_t)(b * SEQ)) * D_MODEL + h * HDIM;
    const _Float16* qr = q + base + (size_t)i * D_MODEL;
    const float q0 = (float)qr[lane];
    const float q1 = (float)qr[lane + 32];

    float m = -1e30f, l = 0.f, c0 = 0.f, c1 = 0.f;

    auto process = [&](int j) {
        const _Float16* kr = k + base + (size_t)j * D_MODEL;
        float p = q0 * (float)kr[lane] + q1 * (float)kr[lane + 32];
#pragma unroll
        for (int off = 16; off > 0; off >>= 1)
            p += __shfl_xor(p, off, 32);
        const float s  = p * 0.125f;               // 1/sqrt(64)
        const float mn = fmaxf(m, s);
        const float sc = __expf(m - mn);
        const float pe = __expf(s - mn);
        const _Float16* vr = v + base + (size_t)j * D_MODEL;
        l  = l  * sc + pe;
        c0 = c0 * sc + pe * (float)vr[lane];
        c1 = c1 * sc + pe * (float)vr[lane + 32];
        m = mn;
    };

    const int blk_start = i & ~(BLK - 1);
    for (int j = BLK - 1; j < blk_start; j += BLK) process(j);  // summary keys
    for (int j = blk_start; j <= i; ++j)          process(j);  // local block

    _Float16* cr = ctx + base + (size_t)i * D_MODEL;
    const float inv_l = 1.0f / l;
    cr[lane]      = (_Float16)(c0 * inv_l);
    cr[lane + 32] = (_Float16)(c1 * inv_l);
}

// ---------------------------------------------------------------------------
// Orchestration
// ---------------------------------------------------------------------------
extern "C" void kernel_launch(void* const* d_in, const int* in_sizes, int n_in,
                              void* d_out, int out_size, void* d_ws, size_t ws_size,
                              hipStream_t stream) {
    const float* x     = (const float*)d_in[0];
    const float* ln1_g = (const float*)d_in[1];
    const float* ln1_b = (const float*)d_in[2];
    const float* wq    = (const float*)d_in[3];
    const float* bq    = (const float*)d_in[4];
    const float* wk    = (const float*)d_in[5];
    const float* bk    = (const float*)d_in[6];
    const float* wv    = (const float*)d_in[7];
    const float* bv    = (const float*)d_in[8];
    const float* wo    = (const float*)d_in[9];
    const float* bo    = (const float*)d_in[10];
    const float* ln2_g = (const float*)d_in[11];
    const float* ln2_b = (const float*)d_in[12];
    const float* w1    = (const float*)d_in[13];
    const float* b1    = (const float*)d_in[14];
    const float* w2    = (const float*)d_in[15];
    const float* b2    = (const float*)d_in[16];
    float* out = (float*)d_out;

    const size_t MB = (size_t)1 << 20;
    char* ws = (char*)d_ws;
    _Float16* wqT = (_Float16*)(ws + 0 * MB);    // [1024][1024]  2MB
    _Float16* wkT = (_Float16*)(ws + 2 * MB);
    _Float16* wvT = (_Float16*)(ws + 4 * MB);
    _Float16* woT = (_Float16*)(ws + 6 * MB);
    _Float16* w1T = (_Float16*)(ws + 8 * MB);    // [4096][1024]  8MB
    _Float16* w2T = (_Float16*)(ws + 16 * MB);   // [1024][4096]  8MB
    _Float16* hbuf = (_Float16*)(ws + 24 * MB);  // [4096][1024]  8MB (h, then h2)
    _Float16* qb   = (_Float16*)(ws + 32 * MB);
    _Float16* kb   = (_Float16*)(ws + 40 * MB);
    _Float16* vb   = (_Float16*)(ws + 48 * MB);
    _Float16* ctxb = (_Float16*)(ws + 56 * MB);
    float*    x1   = (float*)   (ws + 64 * MB);  // fp32 residual  16MB
    _Float16* ff1  = (_Float16*)(ws + 80 * MB);  // [4096][4096]  32MB

    // --- weight convert + transpose (fp32 [K][N] -> f16 [N][K]) ---
    cvt_transpose_kernel<<<(1024 * 1024 + 255) / 256, 256, 0, stream>>>(wq, wqT, 1024, 1024);
    cvt_transpose_kernel<<<(1024 * 1024 + 255) / 256, 256, 0, stream>>>(wk, wkT, 1024, 1024);
    cvt_transpose_kernel<<<(1024 * 1024 + 255) / 256, 256, 0, stream>>>(wv, wvT, 1024, 1024);
    cvt_transpose_kernel<<<(1024 * 1024 + 255) / 256, 256, 0, stream>>>(wo, woT, 1024, 1024);
    cvt_transpose_kernel<<<(1024 * 4096 + 255) / 256, 256, 0, stream>>>(w1, w1T, 1024, 4096);
    cvt_transpose_kernel<<<(4096 * 1024 + 255) / 256, 256, 0, stream>>>(w2, w2T, 4096, 1024);

    // --- LN1 ---
    ln_kernel<<<ROWS, 256, 0, stream>>>(x, ln1_g, ln1_b, hbuf);

    // --- QKV projections (WMMA), f16 out ---
    dim3 gNarrow(D_MODEL / 128, ROWS / 128);   // (8, 32)
    gemm_wmma_kernel<true, false, false, true><<<gNarrow, 256, 0, stream>>>(
        hbuf, wqT, bq, nullptr, nullptr, qb, ROWS, D_MODEL, D_MODEL);
    gemm_wmma_kernel<true, false, false, true><<<gNarrow, 256, 0, stream>>>(
        hbuf, wkT, bk, nullptr, nullptr, kb, ROWS, D_MODEL, D_MODEL);
    gemm_wmma_kernel<true, false, false, true><<<gNarrow, 256, 0, stream>>>(
        hbuf, wvT, bv, nullptr, nullptr, vb, ROWS, D_MODEL, D_MODEL);

    // --- sparse attention ---
    attn_kernel<<<(BATCH * NHEADS * SEQ) / 8, 256, 0, stream>>>(qb, kb, vb, ctxb);

    // --- output projection + residual -> x1 (fp32) ---
    gemm_wmma_kernel<true, false, true, false><<<gNarrow, 256, 0, stream>>>(
        ctxb, woT, bo, x, x1, nullptr, ROWS, D_MODEL, D_MODEL);

    // --- LN2 ---
    ln_kernel<<<ROWS, 256, 0, stream>>>(x1, ln2_g, ln2_b, hbuf);

    // --- FFN up + exact GELU, f16 out ---
    dim3 gWide(HIDDEN / 128, ROWS / 128);      // (32, 32)
    gemm_wmma_kernel<true, true, false, true><<<gWide, 256, 0, stream>>>(
        hbuf, w1T, b1, nullptr, nullptr, ff1, ROWS, HIDDEN, D_MODEL);

    // --- FFN down + residual -> out (fp32) ---
    gemm_wmma_kernel<true, false, true, false><<<gNarrow, 256, 0, stream>>>(
        ff1, w2T, b2, x1, out, nullptr, ROWS, D_MODEL, HIDDEN);
}
